// CustomT5Attention_20736102105339
// MI455X (gfx1250) — compile-verified, matrix operand order checked by reading, not run on hardware
//
#include <hip/hip_runtime.h>
#include <hip/hip_fp16.h>

typedef __attribute__((ext_vector_type(16))) _Float16 v16h;
typedef __attribute__((ext_vector_type(8)))  float    v8f;

#define WMMA_F16(a, b, c) \
  __builtin_amdgcn_wmma_f32_16x16x32_f16(false, (a), false, (b), (short)0, (c), false, false)

__device__ __forceinline__ v8f zero8() {
  v8f z;
#pragma unroll
  for (int i = 0; i < 8; ++i) z[i] = 0.0f;
  return z;
}

// Load a 16-bit WMMA A/B fragment. ISA layout (16-bit A 16x32):
// lane L holds row (base + L%16); dword j holds K-pair
// k = (j/4)*16 + 8*(L/16) + 2*(j%4). Pairs are contiguous in memory,
// and j=0..3 / j=4..7 are each 16B-contiguous (mergeable to b128).
__device__ __forceinline__ v16h load_frag(const _Float16* __restrict__ base,
                                          int row, long long stride,
                                          int colbase, int half) {
  union { v16h h; unsigned u[8]; } r;
  const _Float16* p = base + (long long)row * stride + colbase;
#pragma unroll
  for (int j = 0; j < 8; ++j) {
    int k = ((j >> 2) << 4) + (half << 3) + ((j & 3) << 1);
    r.u[j] = *(const unsigned*)(p + k);
  }
  return r.h;
}

// One async-to-LDS b128 per wave: each lane contributes one 16B transfer.
__device__ __forceinline__ void async_b128(unsigned lds_off, const void* gaddr) {
  asm volatile("global_load_async_to_lds_b128 %0, %1, off"
               :: "v"(lds_off), "v"(gaddr) : "memory");
}
__device__ __forceinline__ unsigned lds_off32(const void* p) {
  return (unsigned)(unsigned long long)p;  // AS3->flat keeps offset in low 32 bits
}

// ---------------- bias bucket table: biasD[h][delta+2047] ----------------
__global__ void bias_table_kernel(const float* __restrict__ emb,
                                  float* __restrict__ biasD) {
  int idx = blockIdx.x * 256 + threadIdx.x;
  if (idx >= 8 * 4095) return;
  int h = idx / 4095;
  int di = idx - h * 4095;
  int delta = di - 2047;                 // delta = k - q (mem - ctx)
  int bucket = (delta > 0) ? 16 : 0;     // num_buckets//2
  int rel = delta < 0 ? -delta : delta;
  if (rel < 8) {                          // max_exact = 8
    bucket += rel;
  } else {
    float v = __logf((float)rel * 0.125f) * (8.0f / __logf(16.0f));
    int vi = 8 + (int)v;                  // trunc toward zero like astype(int32)
    bucket += (vi < 15 ? vi : 15);
  }
  biasD[idx] = emb[bucket * 8 + h];
}

// ---------------- fp32 -> f16 convert ----------------
__global__ void cvt_f16_kernel(const float* __restrict__ in,
                               _Float16* __restrict__ out, int n) {
  int i = blockIdx.x * 256 + threadIdx.x;
  if (i < n) out[i] = (_Float16)in[i];
}

// transpose 512x512 weight to f16: WT[n][k] = W[k][n]
__global__ void cvt_wT_kernel(const float* __restrict__ W,
                              _Float16* __restrict__ WT) {
  int i = blockIdx.x * 256 + threadIdx.x;   // 512*512 threads
  int k = i >> 9, n = i & 511;
  WT[n * 512 + k] = (_Float16)W[i];
}

// ---------------- generic f16 GEMM: [M x 512] @ W(512x512) ----------------
// MODE 0: f16 out [M x 512]
// MODE 1: f16 out transposed per head: Vt[((b*8+h)*64+d)*2048 + s]
// MODE 2: f32 out [M x 512]  (final attn_output -> d_out)
template <int MODE>
__global__ void gemm_f16(const _Float16* __restrict__ A,
                         const _Float16* __restrict__ WT,  // [512 x 512], row n = col n of W
                         void* __restrict__ Out) {
  int tid = threadIdx.x;
  int wave = tid >> 5, lane = tid & 31, half = lane >> 4, l16 = lane & 15;
  int mbase = blockIdx.x * 128 + wave * 16;
  int nbase = blockIdx.y * 64;
  v8f acc[4];
#pragma unroll
  for (int i = 0; i < 4; ++i) acc[i] = zero8();

  for (int k0 = 0; k0 < 512; k0 += 32) {
    v16h a = load_frag(A, mbase + l16, 512, k0, half);
    if (k0 + 32 < 512)
      __builtin_prefetch(A + (long long)(mbase + l16) * 512 + k0 + 32, 0, 0);
#pragma unroll
    for (int ns = 0; ns < 4; ++ns) {
      v16h b = load_frag(WT, nbase + ns * 16 + l16, 512, k0, half);
      acc[ns] = WMMA_F16(a, b, acc[ns]);
    }
  }
#pragma unroll
  for (int ns = 0; ns < 4; ++ns) {
#pragma unroll
    for (int r = 0; r < 8; ++r) {
      int m = mbase + r + 8 * half;        // C layout: m = r + 8*(lane/16)
      int n = nbase + ns * 16 + l16;       // n = lane%16
      float v = acc[ns][r];
      if (MODE == 0) {
        ((_Float16*)Out)[(long long)m * 512 + n] = (_Float16)v;
      } else if (MODE == 1) {
        int b = m >> 11, s = m & 2047, h = n >> 6, d = n & 63;
        ((_Float16*)Out)[(((long long)(b * 8 + h)) * 64 + d) * 2048 + s] = (_Float16)v;
      } else {
        ((float*)Out)[(long long)m * 512 + n] = v;
      }
    }
  }
}

// ---------------- flash attention, one (b,h) per block.y/z ----------------
// K/V tiles are shared by all 8 waves -> stage them in LDS with async
// copies (double buffered, ASYNCcnt-fenced) instead of 8x-redundant VMEM.
__global__ void attn_kernel(const _Float16* __restrict__ Qh,  // [B*S x 512]
                            const _Float16* __restrict__ Kh,  // [B*S x 512]
                            const _Float16* __restrict__ Vt,  // [B*H*64 x 2048]
                            const float* __restrict__ biasD,  // [8 x 4095]
                            _Float16* __restrict__ Of16) {    // [B*S x 512]
  __shared__ float s_bias[4095];
  __shared__ __align__(16) _Float16 s_p[8][16 * 32];
  __shared__ __align__(16) _Float16 s_k[2][32 * 64];   // [klocal][d]
  __shared__ __align__(16) _Float16 s_v[2][64 * 32];   // [d][klocal]

  int h = blockIdx.y, b = blockIdx.z;
  int tid = threadIdx.x;
  int wave = tid >> 5, lane = tid & 31, hlf = lane >> 4, l16 = lane & 15;

  for (int i = tid; i < 4095; i += 256) s_bias[i] = biasD[h * 4095 + i];

  int qbase = blockIdx.x * 128 + wave * 16;
  const _Float16* Qp = Qh + (long long)b * 2048 * 512 + h * 64;
  const _Float16* Kp = Kh + (long long)b * 2048 * 512 + h * 64;
  const _Float16* Vp = Vt + (long long)(b * 8 + h) * 64 * 2048;

  // per-thread slices of the cooperative tile copies
  int krow = tid >> 3, kch = tid & 7;   // K tile: 32 rows x 128B
  int vrow = tid >> 2, vch = tid & 3;   // V tile: 64 rows x 64B

  v16h qa0 = load_frag(Qp, qbase + l16, 512, 0, hlf);
  v16h qa1 = load_frag(Qp, qbase + l16, 512, 32, hlf);

  v8f o[4];
#pragma unroll
  for (int i = 0; i < 4; ++i) o[i] = zero8();
  float mrow[8], lrow[8];
#pragma unroll
  for (int r = 0; r < 8; ++r) { mrow[r] = -1e30f; lrow[r] = 0.0f; }

  // prime buffer 0
  async_b128(lds_off32(&s_k[0][krow * 64 + kch * 8]),
             Kp + (long long)krow * 512 + kch * 8);
  async_b128(lds_off32(&s_v[0][vrow * 32 + vch * 8]),
             Vp + (long long)vrow * 2048 + vch * 8);
  asm volatile("s_wait_asynccnt 0x0" ::: "memory");
  __syncthreads();

  for (int it = 0; it < 64; ++it) {
    int k0 = it * 32;
    int cur = it & 1;
    if (it + 1 < 64) {  // prefetch next tile while computing this one
      int nxt = cur ^ 1;
      async_b128(lds_off32(&s_k[nxt][krow * 64 + kch * 8]),
                 Kp + (long long)(k0 + 32 + krow) * 512 + kch * 8);
      async_b128(lds_off32(&s_v[nxt][vrow * 32 + vch * 8]),
                 Vp + (long long)vrow * 2048 + (k0 + 32) + vch * 8);
    }

    // ---- scores: 16q x 32k, two 16x16 C tiles, K-dim = d (64 -> 2 WMMAs each)
    v8f c0 = zero8(), c1 = zero8();
    {
      v16h bk = load_frag(&s_k[cur][0], l16, 64, 0, hlf);
      c0 = WMMA_F16(qa0, bk, c0);
      bk = load_frag(&s_k[cur][0], l16, 64, 32, hlf);
      c0 = WMMA_F16(qa1, bk, c0);
      bk = load_frag(&s_k[cur][0], 16 + l16, 64, 0, hlf);
      c1 = WMMA_F16(qa0, bk, c1);
      bk = load_frag(&s_k[cur][0], 16 + l16, 64, 32, hlf);
      c1 = WMMA_F16(qa1, bk, c1);
    }
    // ---- add T5 position bias (depends only on k-q and h)
#pragma unroll
    for (int r = 0; r < 8; ++r) {
      int q = qbase + r + 8 * hlf;
      c0[r] += s_bias[k0 + l16 - q + 2047];
      c1[r] += s_bias[k0 + 16 + l16 - q + 2047];
    }
    // ---- online softmax per row; rows live in 16-lane groups, reduce with xor<16
#pragma unroll
    for (int r = 0; r < 8; ++r) {
      float mx = fmaxf(c0[r], c1[r]);
#pragma unroll
      for (int msk = 8; msk >= 1; msk >>= 1)
        mx = fmaxf(mx, __shfl_xor(mx, msk, 32));
      float mnew = fmaxf(mrow[r], mx);
      float alpha = __expf(mrow[r] - mnew);
      mrow[r] = mnew;
      float p0 = __expf(c0[r] - mnew);
      float p1 = __expf(c1[r] - mnew);
      float rs = p0 + p1;
#pragma unroll
      for (int msk = 8; msk >= 1; msk >>= 1)
        rs += __shfl_xor(rs, msk, 32);
      lrow[r] = lrow[r] * alpha + rs;
#pragma unroll
      for (int c = 0; c < 4; ++c) o[c][r] *= alpha;
      // stage P (C-layout element -> row-major 16x32 LDS tile)
      s_p[wave][(r + 8 * hlf) * 32 + l16] = (_Float16)p0;
      s_p[wave][(r + 8 * hlf) * 32 + 16 + l16] = (_Float16)p1;
    }
    asm volatile("s_wait_dscnt 0x0" ::: "memory");
    // ---- reload P in A-fragment layout
    union { v16h hh; unsigned u[8]; } pa;
    {
      const unsigned* p32 = (const unsigned*)(&s_p[wave][0]);
#pragma unroll
      for (int j = 0; j < 8; ++j) {
        int kk = ((j >> 2) << 4) + (hlf << 3) + ((j & 3) << 1);
        pa.u[j] = p32[(l16 * 32 + kk) >> 1];
      }
    }
    // ---- O += P @ V  (B fragments from LDS V tile: [d][klocal])
#pragma unroll
    for (int ds = 0; ds < 4; ++ds) {
      v16h vb = load_frag(&s_v[cur][0], ds * 16 + l16, 32, 0, hlf);
      o[ds] = WMMA_F16(pa.hh, vb, o[ds]);
    }

    // prefetched tile done + everyone finished reading the old one
    asm volatile("s_wait_asynccnt 0x0" ::: "memory");
    __syncthreads();
  }

  // ---- normalize and store f16 [ (b*2048+q)*512 + h*64 + d ]
  long long outbase = (long long)b * 2048 * 512 + h * 64;
#pragma unroll
  for (int r = 0; r < 8; ++r) {
    float inv = 1.0f / lrow[r];
    int q = qbase + r + 8 * hlf;
#pragma unroll
    for (int ds = 0; ds < 4; ++ds)
      Of16[outbase + (long long)q * 512 + ds * 16 + l16] = (_Float16)(o[ds][r] * inv);
  }
}

// ---------------- position_bias output: [1,8,2048,2048], b128 stores ------
__global__ void bias_out_kernel(const float* __restrict__ biasD,
                                float* __restrict__ out) {
  long long idx4 = ((long long)blockIdx.x * 256 + threadIdx.x) * 4;
  int k = (int)(idx4 & 2047);              // multiple of 4, row not crossed
  int q = (int)((idx4 >> 11) & 2047);
  int h = (int)(idx4 >> 22);
  const float* t = biasD + h * 4095 + (k - q + 2047);
  float4 v = make_float4(t[0], t[1], t[2], t[3]);
  *(float4*)(out + idx4) = v;
}

extern "C" void kernel_launch(void* const* d_in, const int* in_sizes, int n_in,
                              void* d_out, int out_size, void* d_ws, size_t ws_size,
                              hipStream_t stream) {
  const float* hidden = (const float*)d_in[0];   // [2,2048,512]
  const float* Wq = (const float*)d_in[1];       // [512,512]
  const float* Wk = (const float*)d_in[2];
  const float* Wv = (const float*)d_in[3];
  const float* Wo = (const float*)d_in[4];
  const float* emb = (const float*)d_in[5];      // [32,8]

  char* ws = (char*)d_ws;
  float*    biasD = (float*)(ws + 0);              // 8*4095*4   = 131040 B
  _Float16* Xh    = (_Float16*)(ws + 131072);      // 4096*512*2 = 4 MB
  _Float16* WqT   = (_Float16*)(ws + 4325376);     // 512 KB each
  _Float16* WkT   = (_Float16*)(ws + 4849664);
  _Float16* WvT   = (_Float16*)(ws + 5373952);
  _Float16* WoT   = (_Float16*)(ws + 5898240);
  _Float16* Qh    = (_Float16*)(ws + 6422528);     // 4 MB
  _Float16* Kh    = (_Float16*)(ws + 10616832);    // 4 MB
  _Float16* Vt    = (_Float16*)(ws + 14811136);    // 4 MB, [B*H*64 x 2048]
  _Float16* Of16  = (_Float16*)(ws + 19005440);    // 4 MB

  float* attn_out = (float*)d_out;                  // [2,2048,512]
  float* bias_out = (float*)d_out + 2LL * 2048 * 512;  // [1,8,2048,2048]

  // 1) per-head bias-by-delta table
  bias_table_kernel<<<(8 * 4095 + 255) / 256, 256, 0, stream>>>(emb, biasD);
  // 2) converts
  cvt_f16_kernel<<<(4096 * 512) / 256, 256, 0, stream>>>(hidden, Xh, 4096 * 512);
  cvt_wT_kernel<<<(512 * 512) / 256, 256, 0, stream>>>(Wq, WqT);
  cvt_wT_kernel<<<(512 * 512) / 256, 256, 0, stream>>>(Wk, WkT);
  cvt_wT_kernel<<<(512 * 512) / 256, 256, 0, stream>>>(Wv, WvT);
  cvt_wT_kernel<<<(512 * 512) / 256, 256, 0, stream>>>(Wo, WoT);
  // 3) projections (M=4096, N=512): grid (M/128, N/64)
  dim3 gg(32, 8);
  gemm_f16<0><<<gg, 256, 0, stream>>>(Xh, WqT, (void*)Qh);
  gemm_f16<0><<<gg, 256, 0, stream>>>(Xh, WkT, (void*)Kh);
  gemm_f16<1><<<gg, 256, 0, stream>>>(Xh, WvT, (void*)Vt);
  // 4) flash attention: grid (S/128, H, B)
  attn_kernel<<<dim3(16, 8, 2), 256, 0, stream>>>(Qh, Kh, Vt, biasD, Of16);
  // 5) output projection straight to d_out (fp32)
  gemm_f16<2><<<gg, 256, 0, stream>>>(Of16, WoT, (void*)attn_out);
  // 6) materialize position_bias output (b128 stores)
  bias_out_kernel<<<(8LL * 2048 * 2048) / (256 * 4), 256, 0, stream>>>(biasD, bias_out);
}
